// AttentionDecoder_14937896255663
// MI455X (gfx1250) — matrix operand order repaired and checked by reading, not compiled, not run on hardware
//
#include <hip/hip_runtime.h>
#include <hip/hip_bf16.h>
#include <math.h>

#define TT 64
#define BB 64
#define LL 2048
#define EE 128
#define HH 256
#define CCH 8   // L-chunks for ctx reduction

typedef __attribute__((ext_vector_type(16))) _Float16 v16h;
typedef __attribute__((ext_vector_type(8)))  float    v8f;

#if defined(__has_builtin)
#if __has_builtin(__builtin_amdgcn_tanhf)
#define FAST_TANH(x) __builtin_amdgcn_tanhf(x)
#endif
#endif
#ifndef FAST_TANH
#define FAST_TANH(x) tanhf(x)
#endif

__device__ __forceinline__ float sigf(float x) { return 1.0f / (1.0f + expf(-x)); }

// ---- WMMA A fragment: 16x32 f16 from row-major (stride lda) ----------------
// lane<16 : row=lane,    halfs[0..15]=K0..15
// lane>=16: row=lane-16, halfs[0..15]=K16..31
__device__ __forceinline__ v16h load_a16(const _Float16* __restrict__ A, int lda) {
  int lane = threadIdx.x & 31;
  int row  = lane & 15;
  int koff = (lane >> 4) << 4;
  const _Float16* p = A + row * lda;
  v16h a;
#pragma unroll
  for (int i = 0; i < 8; ++i) { a[i] = p[koff + i]; a[i + 8] = p[8 + koff + i]; }
  return a;
}

// ---- pack row-major f32 W[K,N] into fragment-major f16 -----------------------
// P[((kt*NT+tn)*32 + lane)*16 + j] = W[k, col]
// lane<16: col=tn*16+lane, K=kt*32 + 0..15 ; lane>=16: K=kt*32 + 16..31
__global__ void k_packB(const float* __restrict__ W, _Float16* __restrict__ P,
                        int K, int N) {
  int e = blockIdx.x * blockDim.x + threadIdx.x;
  if (e >= K * N) return;
  int ncols = N >> 4;
  int tile = e >> 9;      // /512
  int r    = e & 511;
  int lane = r >> 4;
  int j    = r & 15;
  int kt = tile / ncols, tn = tile - kt * ncols;
  int col = (tn << 4) + (lane & 15);
  int k   = (kt << 5) + ((lane >> 4) << 4) + j;
  P[e] = (_Float16)W[k * N + col];
}

// ---- templated WMMA GEMM: C = act(A @ B + bias), B pre-packed ---------------
// ACT: 0=none 1=relu 2=tanh; MIRROR writes an f16 copy into Ch (concat staging)
template <int M, int N, int K, int ACT, bool MIRROR>
__global__ __launch_bounds__(256) void k_gemm(
    const _Float16* __restrict__ A, int lda,
    const _Float16* __restrict__ Bp,
    const float* __restrict__ bias,
    float* __restrict__ C, int ldc,
    _Float16* __restrict__ Ch, int ldch)
{
  constexpr int MT = M >> 4, NT = N >> 4, KT = K >> 5;
  int wave   = blockIdx.x * 8 + (threadIdx.x >> 5);
  int nwaves = gridDim.x * 8;
  int lane   = threadIdx.x & 31;
  int col    = lane & 15;
  int rbase  = (lane >> 4) << 3;
  for (int tile = wave; tile < MT * NT; tile += nwaves) {
    int tm = tile / NT, tn = tile - tm * NT;
    const _Float16* Ap = A + (tm << 4) * lda;
    v8f acc = {};
#pragma unroll
    for (int kt = 0; kt < KT; ++kt) {
      v16h a = load_a16(Ap + (kt << 5), lda);
      v16h b = *(const v16h*)(Bp + ((((size_t)kt * NT + tn) << 5) + lane) * 16);
      acc = __builtin_amdgcn_wmma_f32_16x16x32_f16(false, a, false, b,
                                                   (short)0, acc, false, false);
    }
    int c0 = (tn << 4) + col;
    float bv = bias[c0];
#pragma unroll
    for (int r = 0; r < 8; ++r) {
      float vv = acc[r] + bv;
      if (ACT == 1)      vv = fmaxf(vv, 0.0f);
      else if (ACT == 2) vv = FAST_TANH(vv);
      int row = (tm << 4) + rbase + r;
      C[row * ldc + c0] = vv;
      if (MIRROR) Ch[row * ldch + c0] = (_Float16)vv;
    }
  }
}

// ---- small helper kernels --------------------------------------------------
__global__ void k_init(const float* __restrict__ ic, const float* __restrict__ ih,
                       float* c, float* h, float* cov, float* ctx,
                       _Float16* A1, _Float16* A0) {
  int i = blockIdx.x * blockDim.x + threadIdx.x;
  if (i < BB * LL) cov[i] = 0.0f;
  if (i < BB * HH) {
    int b = i / HH, j = i % HH;
    c[i] = ic[i]; h[i] = ih[i]; ctx[i] = 0.0f;
    A1[b * 384 + 128 + j] = (_Float16)ih[i];
    A0[b * 384 + 128 + j] = (_Float16)0.0f;
  }
}

__global__ void k_inp(const float* __restrict__ inp_t, _Float16* __restrict__ A0) {
  int i = blockIdx.x * blockDim.x + threadIdx.x;
  if (i < BB * EE) { int b = i / EE, e = i % EE; A0[b * 384 + e] = (_Float16)inp_t[i]; }
}

__global__ void k_gates(const float* __restrict__ z, float* __restrict__ c,
                        float* __restrict__ h, _Float16* A1, _Float16* A2, _Float16* A3) {
  int i = blockIdx.x * blockDim.x + threadIdx.x;
  if (i >= BB * HH) return;
  int b = i / HH, j = i % HH;
  const float* zr = z + b * 4 * HH;
  float ig = sigf(zr[j]);
  float fg = sigf(zr[HH + j]);
  float gg = FAST_TANH(zr[2 * HH + j]);
  float og = sigf(zr[3 * HH + j]);
  float cn = fg * c[i] + ig * gg;
  float hn = og * FAST_TANH(cn);
  c[i] = cn; h[i] = hn;
  A2[b * 512 + j]       = (_Float16)cn;
  A2[b * 512 + 256 + j] = (_Float16)hn;
  A1[b * 384 + 128 + j] = (_Float16)hn;   // h for next step's LSTM input
  A3[b * 512 + j]       = (_Float16)hn;   // h for this step's output GEMM
}

// e[b,l] = sum_h v[h]*tanh(enc[b,l,h] + dfeat[b,h] + cov[b,l]*wc[h]); wave per (b,l)
__global__ __launch_bounds__(256) void k_attn_e(const float* __restrict__ enc,
    const float* __restrict__ dfeat, const float* __restrict__ cov,
    const float* __restrict__ v, const float* __restrict__ wc,
    float* __restrict__ e) {
  int wave = blockIdx.x * 8 + (threadIdx.x >> 5);
  int lane = threadIdx.x & 31;
  if (wave >= BB * LL) return;
  int b = wave >> 11;
  float cv = cov[wave];
  const float* ep = enc + (size_t)wave * HH;
  const float* dp = dfeat + b * HH;
  float p = 0.0f;
#pragma unroll
  for (int h = lane; h < HH; h += 32) {
    float f = ep[h] + dp[h] + cv * wc[h];
    p += v[h] * FAST_TANH(f);
  }
#pragma unroll
  for (int off = 16; off > 0; off >>= 1) p += __shfl_xor(p, off);
  if (lane == 0) e[wave] = p;
}

// masked+renormalized softmax over L; writes a, attn output, cov += a
__global__ __launch_bounds__(256) void k_softmax(const float* __restrict__ e,
    const float* __restrict__ mask, float* __restrict__ a,
    float* __restrict__ attn_out, float* __restrict__ cov) {
  __shared__ float red[256];
  int b = blockIdx.x, tid = threadIdx.x;
  const float* eb = e + b * LL;
  float m = -3.4e38f;
  for (int l = tid; l < LL; l += 256) m = fmaxf(m, eb[l]);
  red[tid] = m; __syncthreads();
  for (int s = 128; s > 0; s >>= 1) { if (tid < s) red[tid] = fmaxf(red[tid], red[tid + s]); __syncthreads(); }
  m = red[0]; __syncthreads();
  float q[LL / 256];
  float s = 0.0f;
  int i = 0;
  for (int l = tid; l < LL; l += 256, ++i) {
    float qq = expf(eb[l] - m) * mask[b * LL + l];
    q[i] = qq; s += qq;
  }
  red[tid] = s; __syncthreads();
  for (int st = 128; st > 0; st >>= 1) { if (tid < st) red[tid] += red[tid + st]; __syncthreads(); }
  float inv = 1.0f / red[0];
  i = 0;
  for (int l = tid; l < LL; l += 256, ++i) {
    float av = q[i] * inv;
    a[b * LL + l] = av;
    attn_out[b * LL + l] = av;
    cov[b * LL + l] += av;
  }
}

// ctx two-stage reduction: part[b,ch,h] = sum over l-chunk, then reduce chunks
__global__ __launch_bounds__(256) void k_ctx_part(const float* __restrict__ enc,
    const float* __restrict__ a, float* __restrict__ part) {
  int b = blockIdx.x / CCH, ch = blockIdx.x % CCH;
  int h = threadIdx.x;
  int l0 = ch * (LL / CCH);
  const float* ep = enc + ((size_t)b * LL + l0) * HH + h;
  const float* ab = a + b * LL + l0;
  float acc = 0.0f;
#pragma unroll 4
  for (int l = 0; l < LL / CCH; ++l) acc += ab[l] * ep[(size_t)l * HH];
  part[(b * CCH + ch) * HH + h] = acc;
}

__global__ void k_ctx_reduce(const float* __restrict__ part, float* __restrict__ ctx,
                             _Float16* __restrict__ A0, _Float16* __restrict__ A3) {
  int i = blockIdx.x * blockDim.x + threadIdx.x;
  if (i >= BB * HH) return;
  int b = i / HH, h = i % HH;
  float acc = 0.0f;
#pragma unroll
  for (int ch = 0; ch < CCH; ++ch) acc += part[(b * CCH + ch) * HH + h];
  ctx[i] = acc;
  A0[b * 384 + 128 + h] = (_Float16)acc;   // ctx for next step's x GEMM
  A3[b * 512 + 256 + h] = (_Float16)acc;   // ctx for this step's output GEMM
}

// p_gen[b] = sigmoid([ctx,c,h,x] @ W_p + b_p); one wave per b
__global__ __launch_bounds__(256) void k_pgen(const float* __restrict__ ctx,
    const float* __restrict__ c, const float* __restrict__ h,
    const float* __restrict__ x, const float* __restrict__ Wp,
    const float* __restrict__ bp, float* __restrict__ pg) {
  int wave = blockIdx.x * 8 + (threadIdx.x >> 5);
  int lane = threadIdx.x & 31;
  if (wave >= BB) return;
  int b = wave;
  float p = 0.0f;
  for (int i = lane; i < 3 * HH + EE; i += 32) {
    float val;
    if (i < 256)      val = ctx[b * HH + i];
    else if (i < 512) val = c[b * HH + i - 256];
    else if (i < 768) val = h[b * HH + i - 512];
    else              val = x[b * EE + i - 768];
    p += val * Wp[i];
  }
  for (int off = 16; off > 0; off >>= 1) p += __shfl_xor(p, off);
  if (lane == 0) pg[b] = sigf(p + bp[0]);
}

__global__ void k_final(const float* c, const float* h, const float* cov,
                        float* oc, float* oh, float* ocov) {
  int i = blockIdx.x * blockDim.x + threadIdx.x;
  if (i < BB * LL) ocov[i] = cov[i];
  if (i < BB * HH) { oc[i] = c[i]; oh[i] = h[i]; }
}

// ---- host orchestration ----------------------------------------------------
extern "C" void kernel_launch(void* const* d_in, const int* in_sizes, int n_in,
                              void* d_out, int out_size, void* d_ws, size_t ws_size,
                              hipStream_t stream) {
  const float* dec_in = (const float*)d_in[0];
  const float* init_c = (const float*)d_in[1];
  const float* init_h = (const float*)d_in[2];
  const float* enc    = (const float*)d_in[3];
  const float* mask   = (const float*)d_in[4];
  const float* W_x    = (const float*)d_in[5];
  const float* b_x    = (const float*)d_in[6];
  const float* W_l    = (const float*)d_in[7];
  const float* b_l    = (const float*)d_in[8];
  const float* W_s    = (const float*)d_in[9];
  const float* b_s    = (const float*)d_in[10];
  const float* vv     = (const float*)d_in[11];
  const float* w_c    = (const float*)d_in[12];
  const float* W_p    = (const float*)d_in[13];
  const float* b_p    = (const float*)d_in[14];
  const float* W_o    = (const float*)d_in[15];
  const float* b_o    = (const float*)d_in[16];

  float* out      = (float*)d_out;
  float* out_outs = out;
  float* out_c    = out + (size_t)TT * BB * HH;
  float* out_h    = out_c + BB * HH;
  float* out_attn = out_h + BB * HH;
  float* out_pg   = out_attn + (size_t)TT * BB * LL;
  float* out_cov  = out_pg + TT * BB;

  char* w = (char*)d_ws;
  auto alloc = [&](size_t bytes) -> char* {
    char* p = w; w += (bytes + 255) & ~(size_t)255; return p;
  };
  _Float16* WxP = (_Float16*)alloc(384 * 128 * 2);    // packed fragments
  _Float16* WlP = (_Float16*)alloc(384 * 1024 * 2);
  _Float16* WsP = (_Float16*)alloc(512 * 256 * 2);
  _Float16* WoP = (_Float16*)alloc(512 * 256 * 2);
  _Float16* A0  = (_Float16*)alloc(BB * 384 * 2);     // [inp, ctx]
  _Float16* A1  = (_Float16*)alloc(BB * 384 * 2);     // [x, h]
  _Float16* A2  = (_Float16*)alloc(BB * 512 * 2);     // [c, h]
  _Float16* A3  = (_Float16*)alloc(BB * 512 * 2);     // [h, ctx]
  float* xbuf   = (float*)alloc(BB * EE * 4);
  float* zbuf   = (float*)alloc(BB * 4 * HH * 4);
  float* cbuf   = (float*)alloc(BB * HH * 4);
  float* hbuf   = (float*)alloc(BB * HH * 4);
  float* ctxbuf = (float*)alloc(BB * HH * 4);
  float* dfeat  = (float*)alloc(BB * HH * 4);
  float* cov    = (float*)alloc((size_t)BB * LL * 4);
  float* ebuf   = (float*)alloc((size_t)BB * LL * 4);
  float* abuf   = (float*)alloc((size_t)BB * LL * 4);
  float* cpart  = (float*)alloc((size_t)BB * CCH * HH * 4);

  k_packB<<<(384 * 128 + 255) / 256, 256, 0, stream>>>(W_x, WxP, 384, 128);
  k_packB<<<(384 * 1024 + 255) / 256, 256, 0, stream>>>(W_l, WlP, 384, 1024);
  k_packB<<<(512 * 256 + 255) / 256, 256, 0, stream>>>(W_s, WsP, 512, 256);
  k_packB<<<(512 * 256 + 255) / 256, 256, 0, stream>>>(W_o, WoP, 512, 256);
  k_init<<<(BB * LL + 255) / 256, 256, 0, stream>>>(init_c, init_h, cbuf, hbuf, cov,
                                                    ctxbuf, A1, A0);

  for (int t = 0; t < TT; ++t) {
    k_inp<<<(BB * EE + 255) / 256, 256, 0, stream>>>(dec_in + (size_t)t * BB * EE, A0);
    // x = relu([inp,ctx] @ W_x + b_x)  -> xbuf f32 and A1[:,0:128] f16
    k_gemm<64, 128, 384, 1, true><<<1, 256, 0, stream>>>(A0, 384, WxP, b_x,
                                                         xbuf, 128, A1, 384);
    // z = [x,h] @ W_lstm + b_lstm
    k_gemm<64, 1024, 384, 0, false><<<8, 256, 0, stream>>>(A1, 384, WlP, b_l,
                                                           zbuf, 1024, nullptr, 0);
    k_gates<<<(BB * HH + 255) / 256, 256, 0, stream>>>(zbuf, cbuf, hbuf, A1, A2, A3);
    // dec_feat = tanh([c,h] @ W_s + b_s)
    k_gemm<64, 256, 512, 2, false><<<2, 256, 0, stream>>>(A2, 512, WsP, b_s,
                                                          dfeat, 256, nullptr, 0);
    k_attn_e<<<BB * LL / 8, 256, 0, stream>>>(enc, dfeat, cov, vv, w_c, ebuf);
    k_softmax<<<BB, 256, 0, stream>>>(ebuf, mask, abuf,
                                      out_attn + (size_t)t * BB * LL, cov);
    k_ctx_part<<<BB * CCH, 256, 0, stream>>>(enc, abuf, cpart);
    k_ctx_reduce<<<(BB * HH + 255) / 256, 256, 0, stream>>>(cpart, ctxbuf, A0, A3);
    k_pgen<<<8, 256, 0, stream>>>(ctxbuf, cbuf, hbuf, xbuf, W_p, b_p, out_pg + t * BB);
    // out = [h,ctx] @ W_o + b_o  -> directly into d_out
    k_gemm<64, 256, 512, 0, false><<<2, 256, 0, stream>>>(A3, 512, WoP, b_o,
                                                          out_outs + (size_t)t * BB * HH,
                                                          256, nullptr, 0);
  }
  k_final<<<(BB * LL + 255) / 256, 256, 0, stream>>>(cbuf, hbuf, cov,
                                                     out_c, out_h, out_cov);
}
